// NSAAttention_3126736191972
// MI455X (gfx1250) — matrix operand order repaired and checked by reading, not compiled, not run on hardware
//
#include <hip/hip_runtime.h>
#include <math.h>

// ---- problem constants (match reference) ----
#define SS 2048
#define HH 16
#define DD 64
#define EE 1024
#define STRIDE_C 16
#define LCMP 32
#define NCMP 127
#define NCMP_P 128
#define LSEL 64
#define NSEL 32
#define TOPN 16
#define WIN 512

typedef _Float16 half_t;
typedef __attribute__((ext_vector_type(16))) _Float16 v16h;
typedef __attribute__((ext_vector_type(8)))  float    v8f;

union AV { v16h v; half_t e[16]; };
union CV { v8f  v; float  e[8];  };

__device__ __forceinline__ float redmax16(float v) {
#pragma unroll
  for (int off = 1; off < 16; off <<= 1)
    v = fmaxf(v, __shfl_xor(v, off, 32));
  return v;
}
__device__ __forceinline__ float redsum16(float v) {
#pragma unroll
  for (int off = 1; off < 16; off <<= 1)
    v += __shfl_xor(v, off, 32);
  return v;
}

// ---------------------------------------------------------------------------
// Kernel 1: q (pre-scaled by 1/sqrt(D), exact: power of 2) and k to f16
// row-major [s][h][d]; v to f16 transposed [h][d][s]
// ---------------------------------------------------------------------------
__global__ void prep_qkv(const float* __restrict__ q, const float* __restrict__ k,
                         const float* __restrict__ v,
                         half_t* __restrict__ qh, half_t* __restrict__ kh,
                         half_t* __restrict__ vT) {
  int idx = blockIdx.x * blockDim.x + threadIdx.x;
  if (idx >= SS * HH * DD) return;
  int s  = idx / (HH * DD);
  int hd = idx % (HH * DD);
  qh[idx] = (half_t)(q[idx] * 0.125f);   // fold softmax scale into Q
  kh[idx] = (half_t)k[idx];
  vT[hd * SS + s] = (half_t)v[idx];
}

// ---------------------------------------------------------------------------
// Kernel 2: mean-pool compressed blocks. Kc f16 [c(128 padded)][h][d],
// VcT f16 [h][d][c(128)]. Pad block 127 = zeros (always masked).
// ---------------------------------------------------------------------------
__global__ void prep_cmp(const float* __restrict__ k, const float* __restrict__ v,
                         half_t* __restrict__ Kc, half_t* __restrict__ VcT) {
  int idx = blockIdx.x * blockDim.x + threadIdx.x;
  if (idx >= NCMP_P * HH * DD) return;
  int c  = idx / (HH * DD);
  int hd = idx % (HH * DD);
  float sk = 0.f, sv = 0.f;
  if (c < NCMP) {
    int base = c * STRIDE_C;
    for (int i = 0; i < LCMP; ++i) {
      sk += k[(size_t)(base + i) * (HH * DD) + hd];
      sv += v[(size_t)(base + i) * (HH * DD) + hd];
    }
    sk *= (1.0f / LCMP);
    sv *= (1.0f / LCMP);
  }
  Kc[idx] = (half_t)sk;
  VcT[hd * NCMP_P + c] = (half_t)sv;
}

// ---------------------------------------------------------------------------
// Kernel 3: gating g = sigmoid(x @ Wg + bg), one block (128 thr) per row
// ---------------------------------------------------------------------------
__global__ void gate_kernel(const float* __restrict__ x, const float* __restrict__ Wg,
                            const float* __restrict__ bg, float* __restrict__ g) {
  __shared__ float red[3][128];
  int s = blockIdx.x, tid = threadIdx.x;
  float a0 = 0.f, a1 = 0.f, a2 = 0.f;
  for (int i = tid; i < EE; i += 128) {
    float xv = x[s * EE + i];
    a0 += xv * Wg[i * 3 + 0];
    a1 += xv * Wg[i * 3 + 1];
    a2 += xv * Wg[i * 3 + 2];
  }
  red[0][tid] = a0; red[1][tid] = a1; red[2][tid] = a2;
  __syncthreads();
  for (int st = 64; st > 0; st >>= 1) {
    if (tid < st) {
      red[0][tid] += red[0][tid + st];
      red[1][tid] += red[1][tid + st];
      red[2][tid] += red[2][tid + st];
    }
    __syncthreads();
  }
  if (tid < 3)
    g[s * 3 + tid] = 1.0f / (1.0f + __expf(-(red[tid][0] + bg[tid])));
}

// ---------------------------------------------------------------------------
// Kernel 4: compressed-KV attention + top-k selection mask.
// One wave32 per (head, 16-query tile).
// ---------------------------------------------------------------------------
__global__ void __launch_bounds__(32)
cmp_attn(const half_t* __restrict__ qh, const half_t* __restrict__ Kc,
         const half_t* __restrict__ VcT,
         float* __restrict__ ocmp, unsigned* __restrict__ selmask) {
  __shared__ half_t Pl[16][NCMP_P + 8];
  __shared__ float  psl[16][NSEL];

  const int lane  = threadIdx.x;
  const int nq    = SS / 16;
  const int h     = blockIdx.x / nq;
  const int qbase = (blockIdx.x % nq) * 16;
  const int mrow  = lane & 15;
  const int hi    = lane >> 4;

  // Q tile in WMMA A-layout (two K=32 chunks); q already carries 1/sqrt(D)
  AV aq[2];
#pragma unroll
  for (int kc = 0; kc < 2; ++kc)
#pragma unroll
    for (int j = 0; j < 16; ++j) {
      int d = 32 * kc + 8 * hi + j + (j >= 8 ? 8 : 0);
      aq[kc].e[j] = qh[((size_t)(qbase + mrow) * HH + h) * DD + d];
    }

  // logits = Q @ Kc^T over 128 (padded) compressed blocks: 8 N-tiles x 2 K-chunks
  CV sc[8];
#pragma unroll
  for (int ct = 0; ct < 8; ++ct) {
    v8f acc = {0.f, 0.f, 0.f, 0.f, 0.f, 0.f, 0.f, 0.f};
    int c = ct * 16 + mrow;
#pragma unroll
    for (int kc = 0; kc < 2; ++kc) {
      AV b;
#pragma unroll
      for (int j = 0; j < 16; ++j)
        b.e[j] = Kc[((size_t)c * HH + h) * DD + 32 * kc + 8 * hi + j + (j >= 8 ? 8 : 0)];
      acc = __builtin_amdgcn_wmma_f32_16x16x32_f16(false, aq[kc].v, false, b.v,
                                                   (short)0, acc, false, false);
    }
    sc[ct].v = acc;
  }

  // mask + softmax (full row of 128 columns) -> normalized p_c -> LDS (f16, [m][c])
#pragma unroll
  for (int r = 0; r < 8; ++r) {
    int t = qbase + r + 8 * hi;
    float lmax = -3.0e38f;
#pragma unroll
    for (int ct = 0; ct < 8; ++ct) {
      int c = ct * 16 + mrow;
      float vv = sc[ct].e[r];
      bool valid = (16 * c + 31) <= t;           // fully-past compressed block
      vv = valid ? vv : -3.0e38f;
      sc[ct].e[r] = vv;
      lmax = fmaxf(lmax, vv);
    }
    float mx = redmax16(lmax);
    float lsum = 0.f;
#pragma unroll
    for (int ct = 0; ct < 8; ++ct) {
      float vv = sc[ct].e[r];
      float p = (vv > -1.0e37f) ? __expf(vv - mx) : 0.f;
      sc[ct].e[r] = p;
      lsum += p;
    }
    float ssum = redsum16(lsum);
    float inv = ssum > 0.f ? 1.0f / ssum : 0.f;  // rows t<31: zero (valid_c)
#pragma unroll
    for (int ct = 0; ct < 8; ++ct) {
      float p = sc[ct].e[r] * inv;
      sc[ct].e[r] = p;
      Pl[r + 8 * hi][ct * 16 + mrow] = (half_t)p;
    }
  }
  __syncthreads();

  // selection-block importance (overlap sum) + top-16 mask, lanes 0..15 each own a row
  if (lane < 16) {
    int t = qbase + lane;
#pragma unroll
    for (int n = 0; n < NSEL; ++n) {
      int c0 = 4 * n - 1; if (c0 < 0) c0 = 0;
      int c1 = 4 * n + 3; if (c1 > NCMP_P - 1) c1 = NCMP_P - 1;
      float sv = 0.f;
      for (int c = c0; c <= c1; ++c) sv += (float)Pl[lane][c];
      psl[lane][n] = sv;
    }
    unsigned mask = 0u;
    for (int i = 0; i < NSEL; ++i) {
      float vi = psl[lane][i];
      int rank = 0;
      for (int j = 0; j < NSEL; ++j) {
        float vj = psl[lane][j];
        if (vj > vi || (vj == vi && j < i)) ++rank;  // top_k tie-break: lower index
      }
      if (rank < TOPN) mask |= (1u << i);
    }
    mask |= (1u << (t / LSEL));                      // always include own block
    selmask[(size_t)t * HH + h] = mask;
  }
  __syncthreads();

  // o_cmp = P(16x128) @ Vc(128x64): reload P in A-layout from LDS
  CV oc[4];
#pragma unroll
  for (int n = 0; n < 4; ++n) oc[n].v = (v8f){0.f,0.f,0.f,0.f,0.f,0.f,0.f,0.f};
#pragma unroll
  for (int ab = 0; ab < 4; ++ab) {
    AV pa;
#pragma unroll
    for (int j = 0; j < 16; ++j) {
      int kdim = ab * 32 + 8 * hi + j + (j >= 8 ? 8 : 0);
      pa.e[j] = Pl[mrow][kdim];
    }
#pragma unroll
    for (int nch = 0; nch < 4; ++nch) {
      AV b;
      int d = nch * 16 + mrow;
#pragma unroll
      for (int j = 0; j < 16; ++j)
        b.e[j] = VcT[((size_t)h * DD + d) * NCMP_P + ab * 32 + 8 * hi + j + (j >= 8 ? 8 : 0)];
      oc[nch].v = __builtin_amdgcn_wmma_f32_16x16x32_f16(false, pa.v, false, b.v,
                                                         (short)0, oc[nch].v, false, false);
    }
  }
#pragma unroll
  for (int nch = 0; nch < 4; ++nch)
#pragma unroll
    for (int r = 0; r < 8; ++r) {
      int t = qbase + r + 8 * hi;
      ocmp[((size_t)t * HH + h) * DD + nch * 16 + mrow] = oc[nch].e[r];
    }
}

// ---------------------------------------------------------------------------
// One online-softmax branch update for a 32-key block (SEL: selected blocks;
// !SEL: sliding window). Fully inlined so m/l/acc arrays stay in registers.
// For SEL, the block's selection bit is uniform per row (32-key blocks never
// straddle a 64-token selection block), passed in selrow[].
// ---------------------------------------------------------------------------
template <bool SEL>
__device__ __forceinline__ void branch_update(
    int qbase, int kb, int h, int mrow, int hi,
    const CV st[2], const bool selrow[8],
    float mm[8], float ll[8], CV oo[4],
    const half_t* __restrict__ vT, half_t Pl[16][40]) {
  float pv[2][8];
#pragma unroll
  for (int r = 0; r < 8; ++r) {
    int t = qbase + r + 8 * hi;
    float v0[2];
    float lmax = -3.0e38f;
#pragma unroll
    for (int hb = 0; hb < 2; ++hb) {
      int tk = kb + 16 * hb + mrow;
      bool ok = (tk <= t);
      if (SEL) ok = ok && selrow[r];
      else     ok = ok && ((t - tk) < WIN);
      float vv = ok ? st[hb].e[r] : -3.0e38f;
      v0[hb] = vv;
      lmax = fmaxf(lmax, vv);
    }
    float rmax = redmax16(lmax);
    float mnew = fmaxf(mm[r], rmax);
    float scale = (mnew <= -1.0e37f) ? 1.0f : __expf(mm[r] - mnew);
    float rsum = 0.f;
#pragma unroll
    for (int hb = 0; hb < 2; ++hb) {
      float p = (v0[hb] > -1.0e37f) ? __expf(v0[hb] - mnew) : 0.f;
      pv[hb][r] = p;
      rsum += p;
    }
    rsum = redsum16(rsum);
    ll[r] = ll[r] * scale + rsum;
    mm[r] = mnew;
#pragma unroll
    for (int nch = 0; nch < 4; ++nch) oo[nch].e[r] *= scale;
  }
  __syncthreads();
#pragma unroll
  for (int r = 0; r < 8; ++r)
#pragma unroll
    for (int hb = 0; hb < 2; ++hb)
      Pl[r + 8 * hi][hb * 16 + mrow] = (half_t)pv[hb][r];
  __syncthreads();
  AV pa;
#pragma unroll
  for (int j = 0; j < 16; ++j) {
    int kdim = 8 * hi + j + (j >= 8 ? 8 : 0);
    pa.e[j] = Pl[mrow][kdim];
  }
#pragma unroll
  for (int nch = 0; nch < 4; ++nch) {
    AV b;
    int d = nch * 16 + mrow;
#pragma unroll
    for (int j = 0; j < 16; ++j)
      b.e[j] = vT[((size_t)h * DD + d) * SS + kb + 8 * hi + j + (j >= 8 ? 8 : 0)];
    oo[nch].v = __builtin_amdgcn_wmma_f32_16x16x32_f16(false, pa.v, false, b.v,
                                                       (short)0, oo[nch].v, false, false);
  }
}

// ---------------------------------------------------------------------------
// Kernel 5: main attention (selected + window branches) + gating + output.
// One wave32 per (head, 16-query tile); flash-style over 32-key blocks.
// True NSA sparsity: key blocks touched by neither the window nor any row's
// selected blocks are skipped entirely (wave-uniform test -> no divergence).
// ---------------------------------------------------------------------------
__global__ void __launch_bounds__(32)
main_attn(const half_t* __restrict__ qh, const half_t* __restrict__ kh,
          const half_t* __restrict__ vT,
          const float* __restrict__ ocmp, const unsigned* __restrict__ selmask,
          const float* __restrict__ g, float* __restrict__ out) {
  __shared__ half_t Pl[16][40];

  const int lane  = threadIdx.x;
  const int nq    = SS / 16;
  const int h     = blockIdx.x / nq;
  const int qbase = (blockIdx.x % nq) * 16;
  const int mrow  = lane & 15;
  const int hi    = lane >> 4;

  AV aq[2];
#pragma unroll
  for (int kc = 0; kc < 2; ++kc)
#pragma unroll
    for (int j = 0; j < 16; ++j) {
      int d = 32 * kc + 8 * hi + j + (j >= 8 ? 8 : 0);
      aq[kc].e[j] = qh[((size_t)(qbase + mrow) * HH + h) * DD + d];
    }

  unsigned smk[8];
  float msel[8], lsel[8], mwin[8], lwin[8];
  CV osel[4], owin[4];
#pragma unroll
  for (int r = 0; r < 8; ++r) {
    msel[r] = -3.0e38f; mwin[r] = -3.0e38f;
    lsel[r] = 0.f;      lwin[r] = 0.f;
    smk[r] = selmask[(size_t)(qbase + r + 8 * hi) * HH + h];
  }
#pragma unroll
  for (int n = 0; n < 4; ++n) {
    osel[n].v = (v8f){0.f,0.f,0.f,0.f,0.f,0.f,0.f,0.f};
    owin[n].v = (v8f){0.f,0.f,0.f,0.f,0.f,0.f,0.f,0.f};
  }

  const int kend = qbase + 16;
  for (int kb = 0; kb < kend; kb += 32) {
    // ---- wave-uniform activity tests (NSA sparsity) ----
    const int sb = kb >> 6;                 // selection block of this key block
    bool selrow[8];
    int anybit = 0;
#pragma unroll
    for (int r = 0; r < 8; ++r) {
      selrow[r] = ((smk[r] >> sb) & 1u) != 0u;
      anybit |= (int)selrow[r];
    }
    const bool selAct = __any(anybit);                  // any row selects sb
    const bool winAct = (qbase - kb - 16) < WIN;        // window reaches block
    if (!(selAct || winAct)) continue;

    if (kb + 32 < kend)
      __builtin_prefetch(&kh[((size_t)(kb + 32 + mrow) * HH + h) * DD], 0, 0);

    // shared logits tile: S = Qs(16x64) @ K^T(64x32)  (scale folded into Q)
    CV st[2];
#pragma unroll
    for (int hb = 0; hb < 2; ++hb) {
      v8f acc = {0.f, 0.f, 0.f, 0.f, 0.f, 0.f, 0.f, 0.f};
      int tk = kb + 16 * hb + mrow;
#pragma unroll
      for (int kc = 0; kc < 2; ++kc) {
        AV b;
#pragma unroll
        for (int j = 0; j < 16; ++j)
          b.e[j] = kh[((size_t)tk * HH + h) * DD + 32 * kc + 8 * hi + j + (j >= 8 ? 8 : 0)];
        acc = __builtin_amdgcn_wmma_f32_16x16x32_f16(false, aq[kc].v, false, b.v,
                                                     (short)0, acc, false, false);
      }
      st[hb].v = acc;
    }

    if (selAct)
      branch_update<true >(qbase, kb, h, mrow, hi, st, selrow, msel, lsel, osel, vT, Pl);
    if (winAct)
      branch_update<false>(qbase, kb, h, mrow, hi, st, selrow, mwin, lwin, owin, vT, Pl);
  }

  // finalize: gate the three branches
#pragma unroll
  for (int r = 0; r < 8; ++r) {
    int t = qbase + r + 8 * hi;
    float g0 = g[t * 3 + 0], g1 = g[t * 3 + 1], g2 = g[t * 3 + 2];
    float is = lsel[r] > 0.f ? 1.0f / lsel[r] : 0.f;
    float iw = lwin[r] > 0.f ? 1.0f / lwin[r] : 0.f;
#pragma unroll
    for (int nch = 0; nch < 4; ++nch) {
      int d = nch * 16 + mrow;
      size_t o = ((size_t)t * HH + h) * DD + d;
      out[o] = g0 * ocmp[o] + g1 * osel[nch].e[r] * is + g2 * owin[nch].e[r] * iw;
    }
  }
}

// ---------------------------------------------------------------------------
extern "C" void kernel_launch(void* const* d_in, const int* in_sizes, int n_in,
                              void* d_out, int out_size, void* d_ws, size_t ws_size,
                              hipStream_t stream) {
  const float* q  = (const float*)d_in[0];
  const float* k  = (const float*)d_in[1];
  const float* v  = (const float*)d_in[2];
  const float* x  = (const float*)d_in[3];
  const float* Wg = (const float*)d_in[4];
  const float* bg = (const float*)d_in[5];
  float* out = (float*)d_out;

  // workspace carve-up (~21.7 MB)
  half_t* qh  = (half_t*)d_ws;                 // S*H*D halves
  half_t* kh  = qh + (size_t)SS * HH * DD;
  half_t* vT  = kh + (size_t)SS * HH * DD;     // [h][d][s]
  half_t* Kc  = vT + (size_t)SS * HH * DD;     // [c(128)][h][d]
  half_t* VcT = Kc + (size_t)NCMP_P * HH * DD; // [h][d][c(128)]
  float*  ocmp = (float*)(VcT + (size_t)NCMP_P * HH * DD); // S*H*D floats
  float*  gbuf = ocmp + (size_t)SS * HH * DD;              // S*3
  unsigned* selm = (unsigned*)(gbuf + (size_t)SS * 3);     // S*H

  prep_qkv<<<(SS * HH * DD + 255) / 256, 256, 0, stream>>>(q, k, v, qh, kh, vT);
  prep_cmp<<<(NCMP_P * HH * DD + 255) / 256, 256, 0, stream>>>(k, v, Kc, VcT);
  gate_kernel<<<SS, 128, 0, stream>>>(x, Wg, bg, gbuf);
  cmp_attn<<<HH * (SS / 16), 32, 0, stream>>>(qh, Kc, VcT, ocmp, selm);
  main_attn<<<HH * (SS / 16), 32, 0, stream>>>(qh, kh, vT, ocmp, selm, gbuf, out);
}
